// G3DCrossAttention_38354057953789
// MI455X (gfx1250) — matrix operand order repaired and checked by reading, not compiled, not run on hardware
//
#include <hip/hip_runtime.h>
#include <hip/hip_bf16.h>
#include <math.h>

// ---------------------------------------------------------------------------
// G3DCrossAttention block for MI455X (gfx1250, wave32, WMMA).
// Shapes fixed by the reference: B=2, C=512, L=2048, G=2048, H=8, D=64.
// ---------------------------------------------------------------------------

#define Bc   2
#define Cc   512
#define Lc   2048
#define Gc   2048
#define Hc   8
#define Dc   64
#define Mrows (Lc * Bc)      // 4096 token rows, row index r = i*B + b
#define FF    (4 * Cc)       // 2048

typedef __bf16 bf16_t;
typedef __attribute__((ext_vector_type(8)))  __bf16 bf16x8;
typedef __attribute__((ext_vector_type(16))) __bf16 bf16x16;
typedef __attribute__((ext_vector_type(8)))  float  f32x8;

// ---------- bf16 bit helpers (round-to-nearest-even split) ------------------
__device__ __forceinline__ unsigned short f2bf_rne(float f) {
    unsigned int u = __float_as_uint(f);
    u += 0x7fffu + ((u >> 16) & 1u);
    return (unsigned short)(u >> 16);
}
__device__ __forceinline__ float bf2f(unsigned short h) {
    return __uint_as_float(((unsigned int)h) << 16);
}
__device__ __forceinline__ bf16x16 load_frag(const unsigned short* p0,
                                             const unsigned short* p1) {
    bf16x16 r;
    *reinterpret_cast<bf16x8*>(&r)       = *reinterpret_cast<const bf16x8*>(p0);
    *(reinterpret_cast<bf16x8*>(&r) + 1) = *reinterpret_cast<const bf16x8*>(p1);
    return r;
}

// ---------------------------------------------------------------------------
// K0: seq [B,C,L] -> xseq [(i*B+b)*C + c]   (tiled transpose, coalesced both ways)
// ---------------------------------------------------------------------------
__global__ __launch_bounds__(256)
void g3d_transpose_in(const float* __restrict__ seq, float* __restrict__ xseq) {
    __shared__ float tile[32][33];
    const int b  = blockIdx.z;
    const int c0 = blockIdx.y * 32;
    const int i0 = blockIdx.x * 32;
    const int tx = threadIdx.x;      // 0..31
    const int ty = threadIdx.y;      // 0..7
    #pragma unroll
    for (int p = 0; p < 4; ++p) {
        int c = c0 + ty + p * 8;
        tile[ty + p * 8][tx] = seq[((size_t)b * Cc + c) * Lc + i0 + tx];
    }
    __syncthreads();
    #pragma unroll
    for (int p = 0; p < 4; ++p) {
        int i = i0 + ty + p * 8;
        int c = c0 + tx;
        xseq[((size_t)(i * Bc + b)) * Cc + c] = tile[tx][ty + p * 8];
    }
}

// ---------------------------------------------------------------------------
// K8: o [(i*B+b)*C + n] -> out [b, c, i]  ([B,C,L])
// ---------------------------------------------------------------------------
__global__ __launch_bounds__(256)
void g3d_transpose_out(const float* __restrict__ o, float* __restrict__ out) {
    __shared__ float tile[32][33];
    const int b  = blockIdx.z;
    const int i0 = blockIdx.x * 32;
    const int c0 = blockIdx.y * 32;
    const int tx = threadIdx.x;
    const int ty = threadIdx.y;
    #pragma unroll
    for (int p = 0; p < 4; ++p) {
        int i = i0 + ty + p * 8;
        tile[ty + p * 8][tx] = o[((size_t)(i * Bc + b)) * Cc + c0 + tx];
    }
    __syncthreads();
    #pragma unroll
    for (int p = 0; p < 4; ++p) {
        int c = c0 + ty + p * 8;
        out[((size_t)b * Cc + c) * Lc + i0 + tx] = tile[tx][ty + p * 8];
    }
}

// ---------------------------------------------------------------------------
// K1: rank-1 collapse vectors.
//   Ak[o] = Wk[o,:]·Wg ; Av[o] = Wv[o,:]·Wg ; Bv[o] = Wv[o,:]·bg + bv[o]
// (bk and Wk@bg cancel inside softmax and are never needed.)
// ---------------------------------------------------------------------------
__global__ __launch_bounds__(256)
void g3d_precompute1(const float* __restrict__ Wk, const float* __restrict__ Wv,
                     const float* __restrict__ Wg, const float* __restrict__ bg,
                     const float* __restrict__ bv,
                     float* __restrict__ Ak, float* __restrict__ Av,
                     float* __restrict__ Bv) {
    int c = blockIdx.x * blockDim.x + threadIdx.x;
    if (c >= Cc) return;
    float ak = 0.f, av = 0.f, bb = 0.f;
    for (int j = 0; j < Cc; ++j) {
        float wg = Wg[j];
        ak += Wk[(size_t)c * Cc + j] * wg;
        float wv = Wv[(size_t)c * Cc + j];
        av += wv * wg;
        bb += wv * bg[j];
    }
    Ak[c] = ak; Av[c] = av; Bv[c] = bb + bv[c];
}

// ---------------------------------------------------------------------------
// K2: fold Ak through Wq:  u1[h,c] = scale * sum_d Ak[hD+d]*Wq[hD+d,c]
//                          c1[h]   = scale * sum_d Ak[hD+d]*bq[hD+d]
// so s1[i,b,h] = u1[h,:]·xseq[i,b,:] + c1[h]
// ---------------------------------------------------------------------------
__global__ __launch_bounds__(256)
void g3d_precompute2(const float* __restrict__ Wq, const float* __restrict__ bq,
                     const float* __restrict__ Ak,
                     float* __restrict__ u1, float* __restrict__ c1) {
    const float scale = 0.125f;             // 1/sqrt(D=64)
    int idx = blockIdx.x * blockDim.x + threadIdx.x;   // h*C + c
    if (idx >= Hc * Cc) return;
    int h = idx >> 9, c = idx & (Cc - 1);
    float s = 0.f;
    for (int d = 0; d < Dc; ++d)
        s += Ak[h * Dc + d] * Wq[(size_t)(h * Dc + d) * Cc + c];
    u1[idx] = s * scale;
    if (c == 0) {
        float t = 0.f;
        for (int d = 0; d < Dc; ++d) t += Ak[h * Dc + d] * bq[h * Dc + d];
        c1[h] = t * scale;
    }
}

// ---------------------------------------------------------------------------
// K3: fused attention + residual + LayerNorm1, one block per token row.
// 8 waves = 8 heads. Each wave: s1 dot-product, then online softmax over G
// (lane-strided, shfl combine). Then x = m*Av + Bv + xseq row, LN.
// ---------------------------------------------------------------------------
__global__ __launch_bounds__(256)
void g3d_attn_ln1(const float* __restrict__ xseq, const float* __restrict__ eIn,
                  const float* __restrict__ u1,   const float* __restrict__ c1h,
                  const float* __restrict__ Av,   const float* __restrict__ Bv,
                  const float* __restrict__ g1,   const float* __restrict__ be1,
                  float* __restrict__ y) {
    __shared__ float xrow[Cc];
    __shared__ float erow[Gc];
    __shared__ float mv[Hc];
    __shared__ float red[16];

    const int r    = blockIdx.x;          // token row
    const int b    = r & (Bc - 1);
    const int tid  = threadIdx.x;
    const int lane = tid & 31;
    const int wave = tid >> 5;            // == head

    xrow[tid]       = xseq[(size_t)r * Cc + tid];
    xrow[tid + 256] = xseq[(size_t)r * Cc + tid + 256];
    #pragma unroll
    for (int p = 0; p < Gc / 256; ++p)
        erow[tid + p * 256] = eIn[(size_t)b * Gc + tid + p * 256];
    __syncthreads();

    // s1 for this head
    float s = 0.f;
    for (int c = lane; c < Cc; c += 32) s += u1[wave * Cc + c] * xrow[c];
    #pragma unroll
    for (int o = 16; o > 0; o >>= 1) s += __shfl_xor(s, o, 32);
    const float s1 = s + c1h[wave];

    // online softmax: m = sum_j e_j exp(e_j*s1) / sum_j exp(e_j*s1)
    float mrun = -__builtin_inff(), ssum = 0.f, wsum = 0.f;
    for (int j = lane; j < Gc; j += 32) {
        float e  = erow[j];
        float t  = e * s1;
        float nm = fmaxf(mrun, t);
        float sc = __expf(mrun - nm);
        float et = __expf(t - nm);
        ssum = ssum * sc + et;
        wsum = wsum * sc + e * et;
        mrun = nm;
    }
    #pragma unroll
    for (int o = 16; o > 0; o >>= 1) {
        float mo = __shfl_xor(mrun, o, 32);
        float so = __shfl_xor(ssum, o, 32);
        float wo = __shfl_xor(wsum, o, 32);
        float nm = fmaxf(mrun, mo);
        float a0 = __expf(mrun - nm), a1 = __expf(mo - nm);
        ssum = ssum * a0 + so * a1;
        wsum = wsum * a0 + wo * a1;
        mrun = nm;
    }
    if (lane == 0) mv[wave] = wsum / ssum;
    __syncthreads();

    // attention output + residual, then LN over C=512 (2 elems / thread)
    const int ca = tid, cb = tid + 256;
    float x0 = mv[ca >> 6] * Av[ca] + Bv[ca] + xrow[ca];
    float x1 = mv[cb >> 6] * Av[cb] + Bv[cb] + xrow[cb];
    float ps = x0 + x1, pq = x0 * x0 + x1 * x1;
    #pragma unroll
    for (int o = 16; o > 0; o >>= 1) {
        ps += __shfl_xor(ps, o, 32);
        pq += __shfl_xor(pq, o, 32);
    }
    if (lane == 0) { red[wave] = ps; red[8 + wave] = pq; }
    __syncthreads();
    float ts = 0.f, tq = 0.f;
    #pragma unroll
    for (int w = 0; w < 8; ++w) { ts += red[w]; tq += red[8 + w]; }
    const float mean = ts * (1.f / Cc);
    const float var  = tq * (1.f / Cc) - mean * mean;
    const float rstd = rsqrtf(var + 1e-5f);
    y[(size_t)r * Cc + ca] = (x0 - mean) * rstd * g1[ca] + be1[ca];
    y[(size_t)r * Cc + cb] = (x1 - mean) * rstd * g1[cb] + be1[cb];
}

// ---------------------------------------------------------------------------
// K6: plain LayerNorm over rows of 512 (used for LN2).
// ---------------------------------------------------------------------------
__global__ __launch_bounds__(256)
void g3d_ln512(const float* __restrict__ in, const float* __restrict__ g,
               const float* __restrict__ beta, float* __restrict__ out) {
    __shared__ float red[16];
    const int r = blockIdx.x, tid = threadIdx.x, lane = tid & 31, wave = tid >> 5;
    float x0 = in[(size_t)r * Cc + tid];
    float x1 = in[(size_t)r * Cc + tid + 256];
    float ps = x0 + x1, pq = x0 * x0 + x1 * x1;
    #pragma unroll
    for (int o = 16; o > 0; o >>= 1) {
        ps += __shfl_xor(ps, o, 32);
        pq += __shfl_xor(pq, o, 32);
    }
    if (lane == 0) { red[wave] = ps; red[8 + wave] = pq; }
    __syncthreads();
    float ts = 0.f, tq = 0.f;
    #pragma unroll
    for (int w = 0; w < 8; ++w) { ts += red[w]; tq += red[8 + w]; }
    const float mean = ts * (1.f / Cc);
    const float var  = tq * (1.f / Cc) - mean * mean;
    const float rstd = rsqrtf(var + 1e-5f);
    out[(size_t)r * Cc + tid]       = (x0 - mean) * rstd * g[tid]       + beta[tid];
    out[(size_t)r * Cc + tid + 256] = (x1 - mean) * rstd * g[tid + 256] + beta[tid + 256];
}

// ---------------------------------------------------------------------------
// WMMA GEMM: out[M,N] = epilogue( A[M,K] @ W[N,K]^T + bias (+res) )
// f32 in/out with bf16 hi/lo split (3 WMMAs/tile/K-step ~ f32 accuracy).
// Block = 256 thr = 8 waves; block tile 128x128; wave tile 64x32; K-step 32.
// ---------------------------------------------------------------------------
#define GTM 128
#define GTN 128
#define GTK 32
#define LSTR 40   // LDS row stride in bf16 elems (80B: 16B aligned, bank-skewed)

__global__ __launch_bounds__(256)
void g3d_gemm_bf16s(const float* __restrict__ A, const float* __restrict__ W,
                    const float* __restrict__ bias, const float* __restrict__ res,
                    float* __restrict__ out, int M, int N, int K, int relu) {
    __shared__ __align__(16) unsigned short AsHi[GTM * LSTR];
    __shared__ __align__(16) unsigned short AsLo[GTM * LSTR];
    __shared__ __align__(16) unsigned short WsHi[GTN * LSTR];
    __shared__ __align__(16) unsigned short WsLo[GTN * LSTR];

    const int tid  = threadIdx.x;
    const int lane = tid & 31;
    const int wave = tid >> 5;
    const int wm   = wave >> 2;          // 0..1 -> 64-row slab
    const int wn   = wave & 3;           // 0..3 -> 32-col slab
    const int mBlk = blockIdx.y * GTM;
    const int nBlk = blockIdx.x * GTN;
    const int half = lane >> 4;          // 0|1 (WMMA lane halves)
    const int l16  = lane & 15;

    f32x8 acc[4][2];
    #pragma unroll
    for (int i = 0; i < 4; ++i)
        #pragma unroll
        for (int j = 0; j < 2; ++j)
            #pragma unroll
            for (int e = 0; e < 8; ++e) acc[i][j][e] = 0.f;

    for (int k0 = 0; k0 < K; k0 += GTK) {
        // ---- stage tiles: f32 -> (hi,lo) bf16 into LDS ----
        #pragma unroll
        for (int p = 0; p < 4; ++p) {           // A tile: 128x32 f32
            int f4  = tid + p * 256;
            int row = f4 >> 3;
            int c4  = (f4 & 7) << 2;
            const float4 v = *reinterpret_cast<const float4*>(
                &A[(size_t)(mBlk + row) * K + k0 + c4]);
            float vv[4] = {v.x, v.y, v.z, v.w};
            #pragma unroll
            for (int e = 0; e < 4; ++e) {
                unsigned short hb = f2bf_rne(vv[e]);
                AsHi[row * LSTR + c4 + e] = hb;
                AsLo[row * LSTR + c4 + e] = f2bf_rne(vv[e] - bf2f(hb));
            }
        }
        #pragma unroll
        for (int p = 0; p < 4; ++p) {           // W tile: 128x32 f32
            int f4  = tid + p * 256;
            int row = f4 >> 3;
            int c4  = (f4 & 7) << 2;
            const float4 v = *reinterpret_cast<const float4*>(
                &W[(size_t)(nBlk + row) * K + k0 + c4]);
            float vv[4] = {v.x, v.y, v.z, v.w};
            #pragma unroll
            for (int e = 0; e < 4; ++e) {
                unsigned short hb = f2bf_rne(vv[e]);
                WsHi[row * LSTR + c4 + e] = hb;
                WsLo[row * LSTR + c4 + e] = f2bf_rne(vv[e] - bf2f(hb));
            }
        }
        if (k0 + GTK < K) {                     // prefetch next K-tile (global_prefetch_b8)
            int row = tid >> 1, c16 = (tid & 1) << 4;
            __builtin_prefetch(&A[(size_t)(mBlk + row) * K + k0 + GTK + c16], 0, 0);
            __builtin_prefetch(&W[(size_t)(nBlk + row) * K + k0 + GTK + c16], 0, 0);
        }
        __syncthreads();

        // ---- fragments (documented CDNA5 16-bit layouts) ----
        bf16x16 aHi[4], aLo[4], bHi[2], bLo[2];
        #pragma unroll
        for (int t = 0; t < 4; ++t) {
            int row = wm * 64 + t * 16 + l16;
            const unsigned short* p = &AsHi[row * LSTR];
            aHi[t] = load_frag(p + 8 * half, p + 16 + 8 * half);
            p = &AsLo[row * LSTR];
            aLo[t] = load_frag(p + 8 * half, p + 16 + 8 * half);
        }
        #pragma unroll
        for (int t = 0; t < 2; ++t) {
            int col = wn * 32 + t * 16 + l16;
            const unsigned short* p = &WsHi[col * LSTR];
            bHi[t] = load_frag(p + 8 * half, p + 16 + 8 * half);
            p = &WsLo[col * LSTR];
            bLo[t] = load_frag(p + 8 * half, p + 16 + 8 * half);
        }

        // ---- 3-product bf16 split MAC: hi*hi + hi*lo + lo*hi ----
        #pragma unroll
        for (int tm = 0; tm < 4; ++tm)
            #pragma unroll
            for (int tn = 0; tn < 2; ++tn) {
                acc[tm][tn] = __builtin_amdgcn_wmma_f32_16x16x32_bf16(
                    false, aHi[tm], false, bHi[tn], (short)0, acc[tm][tn], false, false);
                acc[tm][tn] = __builtin_amdgcn_wmma_f32_16x16x32_bf16(
                    false, aHi[tm], false, bLo[tn], (short)0, acc[tm][tn], false, false);
                acc[tm][tn] = __builtin_amdgcn_wmma_f32_16x16x32_bf16(
                    false, aLo[tm], false, bHi[tn], (short)0, acc[tm][tn], false, false);
            }
        __syncthreads();
    }

    // ---- epilogue: bias (+res) (+relu), D layout: row = 8*half + elem ----
    #pragma unroll
    for (int tm = 0; tm < 4; ++tm)
        #pragma unroll
        for (int tn = 0; tn < 2; ++tn) {
            int n = nBlk + wn * 32 + tn * 16 + l16;
            float bb = bias ? bias[n] : 0.f;
            #pragma unroll
            for (int e = 0; e < 8; ++e) {
                int m = mBlk + wm * 64 + tm * 16 + half * 8 + e;
                float v = acc[tm][tn][e] + bb;
                if (res) v += res[(size_t)m * N + n];
                if (relu) v = fmaxf(v, 0.f);
                out[(size_t)m * N + n] = v;
            }
        }
}

// ---------------------------------------------------------------------------
// Host-side orchestration.
// ---------------------------------------------------------------------------
extern "C" void kernel_launch(void* const* d_in, const int* in_sizes, int n_in,
                              void* d_out, int out_size, void* d_ws, size_t ws_size,
                              hipStream_t stream) {
    (void)in_sizes; (void)n_in; (void)out_size; (void)ws_size;
    const float* seq = (const float*)d_in[0];
    const float* eIn = (const float*)d_in[1];
    const float* Wg  = (const float*)d_in[2];
    const float* bg  = (const float*)d_in[3];
    const float* Wq  = (const float*)d_in[4];
    const float* bq  = (const float*)d_in[5];
    const float* Wk  = (const float*)d_in[6];
    /* d_in[7] == bk cancels inside softmax — provably unused */
    const float* Wv  = (const float*)d_in[8];
    const float* bv  = (const float*)d_in[9];
    const float* Wo  = (const float*)d_in[10];
    const float* bo  = (const float*)d_in[11];
    const float* W1  = (const float*)d_in[12];
    const float* b1  = (const float*)d_in[13];
    const float* W2  = (const float*)d_in[14];
    const float* b2  = (const float*)d_in[15];
    const float* g1  = (const float*)d_in[16];
    const float* be1 = (const float*)d_in[17];
    const float* g2  = (const float*)d_in[18];
    const float* be2 = (const float*)d_in[19];
    float* out = (float*)d_out;

    char* ws = (char*)d_ws;
    size_t off = 0;
    auto alloc = [&](size_t bytes) -> float* {
        float* p = (float*)(ws + off);
        off += (bytes + 255) & ~(size_t)255;
        return p;
    };
    float* xseq = alloc((size_t)Mrows * Cc * 4);   //  8 MB
    float* y    = alloc((size_t)Mrows * Cc * 4);   //  8 MB
    float* h1   = alloc((size_t)Mrows * FF * 4);   // 32 MB
    float* t2   = alloc((size_t)Mrows * Cc * 4);   //  8 MB
    float* z    = alloc((size_t)Mrows * Cc * 4);   //  8 MB
    float* obuf = alloc((size_t)Mrows * Cc * 4);   //  8 MB
    float* Ak   = alloc(Cc * 4);
    float* Av   = alloc(Cc * 4);
    float* Bv   = alloc(Cc * 4);
    float* u1   = alloc(Hc * Cc * 4);
    float* c1   = alloc(Hc * 4);

    g3d_transpose_in<<<dim3(Lc / 32, Cc / 32, Bc), dim3(32, 8), 0, stream>>>(seq, xseq);
    g3d_precompute1<<<(Cc + 255) / 256, 256, 0, stream>>>(Wk, Wv, Wg, bg, bv, Ak, Av, Bv);
    g3d_precompute2<<<(Hc * Cc + 255) / 256, 256, 0, stream>>>(Wq, bq, Ak, u1, c1);
    g3d_attn_ln1<<<Mrows, 256, 0, stream>>>(xseq, eIn, u1, c1, Av, Bv, g1, be1, y);
    g3d_gemm_bf16s<<<dim3(FF / GTN, Mrows / GTM), 256, 0, stream>>>(
        y, W1, b1, nullptr, h1, Mrows, FF, Cc, 1);
    g3d_gemm_bf16s<<<dim3(Cc / GTN, Mrows / GTM), 256, 0, stream>>>(
        h1, W2, b2, y, t2, Mrows, Cc, FF, 0);
    g3d_ln512<<<Mrows, 256, 0, stream>>>(t2, g2, be2, z);
    g3d_gemm_bf16s<<<dim3(Cc / GTN, Mrows / GTM), 256, 0, stream>>>(
        z, Wo, bo, nullptr, obuf, Mrows, Cc, Cc, 0);
    g3d_transpose_out<<<dim3(Lc / 32, Cc / 32, Bc), dim3(32, 8), 0, stream>>>(obuf, out);
}
